// Seq2SeqDecoder_22797686408100
// MI455X (gfx1250) — matrix operand order repaired and checked by reading, not compiled, not run on hardware
//
#include <hip/hip_runtime.h>

typedef __bf16 bf16_t;
typedef __attribute__((ext_vector_type(16))) __bf16 v16bf;
typedef __attribute__((ext_vector_type(8)))  __bf16 v8bf;
typedef __attribute__((ext_vector_type(8)))  float  v8f;

static constexpr int Bsz = 64, Ssz = 256, Dsz = 1024, Hsz = 1024, Esz = 128, Osz = 512;
static constexpr int G4  = 4 * Hsz;   // 4096 gate columns
static constexpr int NBLK = 32;       // persistent grid (32 blocks * 8 waves = 256 gate tiles)

// LDS staging strides (elements), padded +16B/row to spread bank groups
static constexpr int HLD = Hsz + 8;   // 1032 elems = 2064B per row
static constexpr int ELD = Esz + 8;   // 136 elems  = 272B per row
static constexpr int SMEM_P1 = 16 * HLD * 2 + 16 * ELD * 2;   // 37376 B
static constexpr int SMEM_P2 = 16 * Osz * 4 + 64;             // 32832 B
static constexpr int SMEM_BYTES = SMEM_P1 > SMEM_P2 ? SMEM_P1 : SMEM_P2;

// ---------------- CDNA5 WMMA bf16 16x16x32 fragment helpers (wave32) -------
// A (16x32, MxK): lane L -> M = L&15, half = L>>4.
__device__ __forceinline__ v16bf load_a_frag(const bf16_t* row, int k0, int half) {
  v8bf lo = *(const v8bf*)(row + k0 + half * 16);
  v8bf hi = *(const v8bf*)(row + k0 + half * 16 + 8);
  v16bf f;
#pragma unroll
  for (int i = 0; i < 8; ++i) { f[i] = lo[i]; f[i + 8] = hi[i]; }
  return f;
}
// B (32x16, KxN), B[k][n] = W[n][k]: lane L -> N = n0 + (L&15); 16 contiguous bf16.
__device__ __forceinline__ v16bf load_b_frag(const bf16_t* W, int ldw, int n0, int k0, int lane) {
  int n = n0 + (lane & 15);
  int half = lane >> 4;
  return *(const v16bf*)(W + (size_t)n * ldw + k0 + half * 16);
}
__device__ __forceinline__ v8f wmma_bf16(v16bf a, v16bf b, v8f c) {
  return __builtin_amdgcn_wmma_f32_16x16x32_bf16(false, a, false, b, (short)0, c, false, false);
}
// C/D: lane L, vgpr v -> (M = v + (L>>4)*8, N = L&15)

__device__ __forceinline__ float fast_sigmoid(float x) {
  return __fdividef(1.f, 1.f + __expf(-x));
}
__device__ __forceinline__ float fast_tanh(float x) {
  x = fminf(fmaxf(x, -15.f), 15.f);
  float e = __expf(2.f * x);
  return __fdividef(e - 1.f, e + 1.f);
}

// -------- gfx1250 async global->LDS copy (ASYNCcnt-tracked, per-lane 16B) --
__device__ __forceinline__ void async_ld_b128(unsigned lds_off, const void* gptr) {
  asm volatile("global_load_async_to_lds_b128 %0, %1, off"
               :: "v"(lds_off), "v"(gptr) : "memory");
}
__device__ __forceinline__ void wait_async0() {
  asm volatile("s_wait_asynccnt 0" ::: "memory");
}

// ---------------- sense-reversing grid barrier -----------------------------
__device__ __forceinline__ void grid_barrier(unsigned* bar, unsigned nblk) {
  __syncthreads();
  if (threadIdx.x == 0) {
    __threadfence();
    volatile unsigned* vgen = bar + 1;
    unsigned gen = *vgen;
    if (atomicAdd(bar, 1u) == nblk - 1) {
      *bar = 0;
      __threadfence();
      atomicAdd(bar + 1, 1u);
    } else {
      while (*vgen == gen) __builtin_amdgcn_s_sleep(1);
    }
    __threadfence();
  }
  __syncthreads();
}

// ---------------- prep kernels --------------------------------------------
__global__ void k_cast_bf16(const float* __restrict__ src, bf16_t* __restrict__ dst, long n) {
  long i = (long)blockIdx.x * blockDim.x + threadIdx.x;
  long stride = (long)gridDim.x * blockDim.x;
  for (; i < n; i += stride) dst[i] = (bf16_t)src[i];
}

__global__ void k_split_wih(const float* __restrict__ w_ih,
                            bf16_t* __restrict__ Wx, bf16_t* __restrict__ We) {
  const long total = (long)G4 * (Dsz + Esz);
  long i = (long)blockIdx.x * blockDim.x + threadIdx.x;
  long stride = (long)gridDim.x * blockDim.x;
  for (; i < total; i += stride) {
    long r = i / (Dsz + Esz), k = i % (Dsz + Esz);
    float v = w_ih[i];
    if (k < Dsz) Wx[r * Dsz + k] = (bf16_t)v;
    else         We[r * Esz + (k - Dsz)] = (bf16_t)v;
  }
}

__global__ void k_init_state(const float* __restrict__ init_tensor,
                             bf16_t* hA, bf16_t* hB, float* c,
                             bf16_t* emb, unsigned* bar) {
  int i = blockIdx.x * blockDim.x + threadIdx.x;
  int stride = gridDim.x * blockDim.x;
  for (int t = i; t < Bsz * Hsz; t += stride) {
    hA[t] = (bf16_t)0.f; hB[t] = (bf16_t)0.f; c[t] = 0.f;
  }
  for (int t = i; t < Bsz * Esz; t += stride)
    emb[t] = (bf16_t)init_tensor[t & (Esz - 1)];
  if (i < 8) bar[i] = 0;
}

// ------ hoisted GEMM: bias + x @ Wx.T, stored lane-packed per WMMA tile ----
// Xg layout: tile (s, mtb, g, jt) -> base = (((s*4+mtb)*4+g)*64+jt)*256, then
// lane*8 + v holds C(m = mtb*16 + v + (lane>>4)*8, n = g*1024 + jt*16 + (lane&15)).
__global__ void __launch_bounds__(256) k_precompute(
    const bf16_t* __restrict__ Xbf, const bf16_t* __restrict__ Wx,
    const float* __restrict__ b_ih, const float* __restrict__ b_hh,
    bf16_t* __restrict__ Xg) {
  const int wid = threadIdx.x >> 5, lane = threadIdx.x & 31, half = lane >> 4;
  const int wg = blockIdx.x * 8 + wid;    // 0..65535
  const int mt = wg >> 6;                 // 1024 M-tiles over m = s*64+b
  const int ng = wg & 63;                 // 64 groups of 4 N-tiles
  const int m0 = mt * 16;
  const int mrow = m0 + (lane & 15);
  // hidden layout [B][S][D]: row (s,b) at (b*S + s)*D
  const bf16_t* arow = Xbf + ((size_t)(mrow & 63) * Ssz + (mrow >> 6)) * Dsz;

  v8f acc[4];
#pragma unroll
  for (int i = 0; i < 4; ++i) {
    int n = ng * 64 + i * 16 + (lane & 15);
    float bias = b_ih[n] + b_hh[n];
#pragma unroll
    for (int v = 0; v < 8; ++v) acc[i][v] = bias;
  }

  // software-pipelined K loop: B frags for k+1 issued before WMMAs of k
  v16bf bcur[4], bnxt[4];
#pragma unroll
  for (int i = 0; i < 4; ++i)
    bcur[i] = load_b_frag(Wx, Dsz, ng * 64 + i * 16, 0, lane);
#pragma unroll 2
  for (int k0 = 0; k0 < Dsz; k0 += 32) {
    int kn = (k0 + 32 < Dsz) ? k0 + 32 : 0;
#pragma unroll
    for (int i = 0; i < 4; ++i)
      bnxt[i] = load_b_frag(Wx, Dsz, ng * 64 + i * 16, kn, lane);
    v16bf a = load_a_frag(arow, k0, half);
#pragma unroll
    for (int i = 0; i < 4; ++i) acc[i] = wmma_bf16(a, bcur[i], acc[i]);
#pragma unroll
    for (int i = 0; i < 4; ++i) bcur[i] = bnxt[i];
  }

#pragma unroll
  for (int i = 0; i < 4; ++i) {
    int n0 = ng * 64 + i * 16;
    int g = n0 >> 10, jt = (n0 >> 4) & 63;
    v8bf pk;
#pragma unroll
    for (int v = 0; v < 8; ++v) pk[v] = (bf16_t)acc[i][v];
    size_t base = ((((size_t)(m0 >> 6) * 4 + ((m0 >> 4) & 3)) * 4 + g) * 64 + jt) * 256;
    *(v8bf*)(Xg + base + lane * 8) = pk;
  }
}

// ---------------- persistent serial decode loop ----------------------------
__global__ void __launch_bounds__(256) k_decode(
    const bf16_t* __restrict__ Xg, const bf16_t* __restrict__ We,
    const bf16_t* __restrict__ Whh, const bf16_t* __restrict__ Wout,
    const bf16_t* __restrict__ Embt, const float* __restrict__ b_out,
    const int* __restrict__ seq_lens,
    bf16_t* hA, bf16_t* hB, float* __restrict__ c_st, bf16_t* emb,
    unsigned* bar, float* __restrict__ out) {
  __shared__ __align__(16) char smem[SMEM_BYTES];
  bf16_t* s_h   = (bf16_t*)smem;                       // phase 1: 16 x HLD
  bf16_t* s_e   = (bf16_t*)(smem + 16 * HLD * 2);      // phase 1: 16 x ELD
  float*  s_log = (float*)smem;                        // phase 2: 16 x Osz
  int*    s_best = (int*)(smem + 16 * Osz * 4);        // phase 2

  const int wid = threadIdx.x >> 5, lane = threadIdx.x & 31, half = lane >> 4;
  const int tile = blockIdx.x * 8 + wid;   // 0..255 gate tiles
  const int mt = tile >> 6, jt = tile & 63;
  const int j0 = jt * 16;
  const int m0w = (blockIdx.x >> 3) * 16;  // block-uniform batch tile (== mt*16)
  const int jcol = j0 + (lane & 15);
  const bf16_t* arow_h = s_h + (lane & 15) * HLD;
  const bf16_t* arow_e = s_e + (lane & 15) * ELD;

  for (int s = 0; s < Ssz; ++s) {
    const bf16_t* h_in  = (s & 1) ? hB : hA;
    bf16_t*       h_out = (s & 1) ? hA : hB;

    // ---- stage this block's 16 h rows (32KB) + 16 emb rows (4KB) into LDS
    {
      int r = threadIdx.x >> 4, ch = threadIdx.x & 15;   // 16 rows x 16 chunks
      unsigned loff = (unsigned)(size_t)s_h + r * (HLD * 2) + ch * 128;
      const char* gp = (const char*)(h_in + (size_t)m0w * Hsz) + r * 2048 + ch * 128;
#pragma unroll
      for (int q = 0; q < 8; ++q) async_ld_b128(loff + q * 16, gp + q * 16);
      if (threadIdx.x < 32) {                            // 16 rows x 2 chunks
        int r2 = threadIdx.x >> 1, c2 = threadIdx.x & 1;
        unsigned le = (unsigned)(size_t)s_e + r2 * (ELD * 2) + c2 * 128;
        const char* ge = (const char*)(emb + (size_t)(m0w + r2) * Esz) + c2 * 128;
#pragma unroll
        for (int q = 0; q < 8; ++q) async_ld_b128(le + q * 16, ge + q * 16);
      }
    }

    // ---- init accumulators from lane-packed Xg (one b128 per gate), overlaps
    v8f acc[4];
    {
      const bf16_t* xgb = Xg + (((size_t)s * 4 + mt) * 4 * 64 + jt) * 256 + lane * 8;
#pragma unroll
      for (int g = 0; g < 4; ++g) {
        v8bf pk = *(const v8bf*)(xgb + (size_t)g * 64 * 256);
#pragma unroll
        for (int v = 0; v < 8; ++v) acc[g][v] = (float)pk[v];
      }
    }
    wait_async0();
    __syncthreads();

    // ---- emb @ We.T (A from LDS)
#pragma unroll
    for (int k0 = 0; k0 < Esz; k0 += 32) {
      v16bf a = load_a_frag(arow_e, k0, half);
#pragma unroll
      for (int g = 0; g < 4; ++g)
        acc[g] = wmma_bf16(a, load_b_frag(We, Esz, g * Hsz + j0, k0, lane), acc[g]);
    }
    // ---- h @ Whh.T (A from LDS, B double-buffered from L2)
    {
      v16bf bcur[4], bnxt[4];
#pragma unroll
      for (int g = 0; g < 4; ++g)
        bcur[g] = load_b_frag(Whh, Hsz, g * Hsz + j0, 0, lane);
#pragma unroll 2
      for (int k0 = 0; k0 < Hsz; k0 += 32) {
        int kn = (k0 + 32 < Hsz) ? k0 + 32 : 0;
#pragma unroll
        for (int g = 0; g < 4; ++g)
          bnxt[g] = load_b_frag(Whh, Hsz, g * Hsz + j0, kn, lane);
        v16bf a = load_a_frag(arow_h, k0, half);
#pragma unroll
        for (int g = 0; g < 4; ++g) acc[g] = wmma_bf16(a, bcur[g], acc[g]);
#pragma unroll
        for (int g = 0; g < 4; ++g) bcur[g] = bnxt[g];
      }
    }
    // ---- LSTM cell: all 4 gates for (m,j) live in this lane
#pragma unroll
    for (int v = 0; v < 8; ++v) {
      int m = m0w + v + half * 8;
      size_t idx = (size_t)m * Hsz + jcol;
      float ig = fast_sigmoid(acc[0][v]);
      float fg = fast_sigmoid(acc[1][v]);
      float gg = fast_tanh(acc[2][v]);
      float og = fast_sigmoid(acc[3][v]);
      float cn = fg * c_st[idx] + ig * gg;
      c_st[idx] = cn;
      h_out[idx] = (bf16_t)(og * fast_tanh(cn));
    }

    grid_barrier(bar, NBLK);

    // ---- phase 2: logits + argmax + embedding feedback (blocks 0..3)
    if (blockIdx.x < 4) {
      const int m0 = blockIdx.x * 16;
      const bf16_t* arow = h_out + (size_t)(m0 + (lane & 15)) * Hsz;
      v8f acc2[4];
#pragma unroll
      for (int i = 0; i < 4; ++i) {
        float bias = b_out[(wid * 4 + i) * 16 + (lane & 15)];
#pragma unroll
        for (int v = 0; v < 8; ++v) acc2[i][v] = bias;
      }
      v16bf bcur[4], bnxt[4];
#pragma unroll
      for (int i = 0; i < 4; ++i)
        bcur[i] = load_b_frag(Wout, Hsz, (wid * 4 + i) * 16, 0, lane);
#pragma unroll 2
      for (int k0 = 0; k0 < Hsz; k0 += 32) {
        int kn = (k0 + 32 < Hsz) ? k0 + 32 : 0;
#pragma unroll
        for (int i = 0; i < 4; ++i)
          bnxt[i] = load_b_frag(Wout, Hsz, (wid * 4 + i) * 16, kn, lane);
        v16bf a = load_a_frag(arow, k0, half);
#pragma unroll
        for (int i = 0; i < 4; ++i) acc2[i] = wmma_bf16(a, bcur[i], acc2[i]);
#pragma unroll
        for (int i = 0; i < 4; ++i) bcur[i] = bnxt[i];
      }
#pragma unroll
      for (int i = 0; i < 4; ++i) {
        int n = (wid * 4 + i) * 16 + (lane & 15);
#pragma unroll
        for (int v = 0; v < 8; ++v) {
          int mr = v + half * 8;
          float val = acc2[i][v];
          s_log[mr * Osz + n] = val;
          int b = m0 + mr;
          out[((size_t)b * Ssz + s) * Osz + n] = (s < seq_lens[b]) ? val : 0.f;
        }
      }
      __syncthreads();
      if (threadIdx.x < 16) {                 // first-max argmax, like jnp
        const float* row = s_log + threadIdx.x * Osz;
        float bv = row[0]; int bi = 0;
        for (int o = 1; o < Osz; ++o) { float v = row[o]; if (v > bv) { bv = v; bi = o; } }
        s_best[threadIdx.x] = bi;
      }
      __syncthreads();
      for (int t = threadIdx.x; t < 16 * Esz; t += 256) {
        int r = t >> 7, e = t & (Esz - 1);
        emb[(size_t)(m0 + r) * Esz + e] = Embt[(size_t)s_best[r] * Esz + e];
      }
    }

    grid_barrier(bar, NBLK);
  }
}

// ---------------- launch ---------------------------------------------------
extern "C" void kernel_launch(void* const* d_in, const int* in_sizes, int n_in,
                              void* d_out, int out_size, void* d_ws, size_t ws_size,
                              hipStream_t stream) {
  const float* hidden      = (const float*)d_in[0];
  const float* init_tensor = (const float*)d_in[1];
  const float* emb_table   = (const float*)d_in[2];
  const float* w_ih        = (const float*)d_in[3];
  const float* w_hh        = (const float*)d_in[4];
  const float* b_ih        = (const float*)d_in[5];
  const float* b_hh        = (const float*)d_in[6];
  const float* w_out       = (const float*)d_in[7];
  const float* b_out       = (const float*)d_in[8];
  const int*   seq_lens    = (const int*)d_in[9];

  size_t off = 0;
  auto take = [&](size_t bytes) -> void* {
    void* p = (char*)d_ws + off;
    off = (off + bytes + 255) & ~(size_t)255;
    return p;
  };
  bf16_t*  Xbf  = (bf16_t*) take((size_t)Bsz * Ssz * Dsz * 2);   // 33.6 MB
  bf16_t*  Xg   = (bf16_t*) take((size_t)Bsz * Ssz * G4  * 2);   // 134 MB
  bf16_t*  Wx   = (bf16_t*) take((size_t)G4  * Dsz * 2);         // 8 MB
  bf16_t*  We   = (bf16_t*) take((size_t)G4  * Esz * 2);         // 1 MB
  bf16_t*  Whh  = (bf16_t*) take((size_t)G4  * Hsz * 2);         // 8 MB
  bf16_t*  Wout = (bf16_t*) take((size_t)Osz * Hsz * 2);         // 1 MB
  bf16_t*  Embt = (bf16_t*) take((size_t)Osz * Esz * 2);
  bf16_t*  hA   = (bf16_t*) take((size_t)Bsz * Hsz * 2);
  bf16_t*  hB   = (bf16_t*) take((size_t)Bsz * Hsz * 2);
  float*   cst  = (float*)  take((size_t)Bsz * Hsz * 4);
  bf16_t*  emb  = (bf16_t*) take((size_t)Bsz * Esz * 2);
  unsigned* bar = (unsigned*)take(64);

  k_cast_bf16<<<2048, 256, 0, stream>>>(hidden, Xbf, (long)Bsz * Ssz * Dsz);
  k_cast_bf16<<<256,  256, 0, stream>>>(w_hh, Whh, (long)G4 * Hsz);
  k_cast_bf16<<<64,   256, 0, stream>>>(w_out, Wout, (long)Osz * Hsz);
  k_cast_bf16<<<16,   256, 0, stream>>>(emb_table, Embt, (long)Osz * Esz);
  k_split_wih<<<512,  256, 0, stream>>>(w_ih, Wx, We);
  k_init_state<<<64,  256, 0, stream>>>(init_tensor, hA, hB, cst, emb, bar);
  k_precompute<<<8192, 256, 0, stream>>>(Xbf, Wx, b_ih, b_hh, Xg);
  k_decode<<<NBLK, 256, 0, stream>>>(Xg, We, Whh, Wout, Embt, b_out, seq_lens,
                                     hA, hB, cst, emb, bar, (float*)d_out);
}
